// LSTM_54984171323414
// MI455X (gfx1250) — matrix operand order repaired and checked by reading, not compiled, not run on hardware
//
#include <hip/hip_runtime.h>
#include <hip/hip_bf16.h>
#include <math.h>

typedef __bf16 bf16_t;
typedef __attribute__((ext_vector_type(16))) __bf16 v16bf;
typedef __attribute__((ext_vector_type(8)))  __bf16 v8bf;
typedef __attribute__((ext_vector_type(8)))  float  v8f;

// ---- WMMA A-fragment loader: 16-bit A 16x32 layout (CDNA5 ISA) ----
// lane L (<16): row M=L,  elems 0..7 = K 0..7,  elems 8..15 = K 16..23
// lane L (>=16): row M=L-16, elems 0..7 = K 8..15, elems 8..15 = K 24..31
// caller passes p = &A[row*ldA + k0 + (lane>=16 ? 8 : 0)]
static __device__ __forceinline__ v16bf wmma_load_a(const bf16_t* __restrict__ p) {
  const v8bf lo = *(const v8bf*)(p);
  const v8bf hi = *(const v8bf*)(p + 16);
  v16bf a;
#pragma unroll
  for (int e = 0; e < 8; ++e) { a[e] = lo[e]; a[e + 8] = hi[e]; }
  return a;
}

// branch-free sigmoid / tanh on hardware v_exp_f32 (overflow-safe, no EXEC divergence)
static __device__ __forceinline__ float fast_sigmoid(float x) {
  return 1.f / (1.f + __expf(-x));
}
static __device__ __forceinline__ float fast_tanh(float x) {
  const float ax = fabsf(x);
  const float t  = 1.f - 2.f / (__expf(2.f * ax) + 1.f);   // exp overflow -> inf -> t=1
  return copysignf(t, x);
}

// ---- elementwise fp32 -> bf16 ----
__global__ __launch_bounds__(256) void f2bf_kernel(const float* __restrict__ in,
                                                   bf16_t* __restrict__ out, int n) {
  int i = blockIdx.x * 256 + threadIdx.x;
  if (i < n) out[i] = (bf16_t)in[i];
}

// ---- concat 4 gate biases into [4096] ----
__global__ __launch_bounds__(256) void concat_bias_kernel(const float* __restrict__ b0,
    const float* __restrict__ b1, const float* __restrict__ b2, const float* __restrict__ b3,
    float* __restrict__ out) {
  int n = blockIdx.x * 256 + threadIdx.x;   // 0..4095
  int g = n >> 10, c = n & 1023;
  const float* s = (g == 0) ? b0 : (g == 1) ? b1 : (g == 2) ? b2 : b3;
  out[n] = s[c];
}

// ---- pack fp32 row-major [K x 1024] matrices (up to 4, column-concatenated to N)
// into per-lane WMMA B-fragment layout. Tile (kt,nt) is 32x16 (KxN):
//   tileIdx = nt*(K/32) + kt ; storage: dst[tileIdx*512 + lane*16 + j]
//   element  = W[kt*32 + (lane>=16?8:0) + (j&7) + (j>=8?16:0)][nt*16 + (lane&15)]
__global__ __launch_bounds__(256) void pack_b_kernel(const float* __restrict__ s0,
    const float* __restrict__ s1, const float* __restrict__ s2, const float* __restrict__ s3,
    bf16_t* __restrict__ dst, int K, int N) {
  int idx = blockIdx.x * 256 + threadIdx.x;     // grid sized exactly K*N/256
  int tile   = idx >> 9;
  int within = idx & 511;
  int lane = within >> 4;
  int j    = within & 15;
  int KT = K >> 5;
  int nt = tile / KT;
  int kt = tile - nt * KT;
  int k = (kt << 5) + ((lane >> 4) << 3) + (j & 7) + ((j >> 3) << 4);
  int n = (nt << 4) + (lane & 15);
  int g = n >> 10;
  int c = n & 1023;
  const float* s = (g == 0) ? s0 : (g == 1) ? s1 : (g == 2) ? s2 : s3;
  dst[idx] = (bf16_t)s[(size_t)k * 1024 + c];
}

// ---- bf16 WMMA GEMM: out[M,N] = A[M,K] @ packedB + bias[N] ----
// One wave computes a 16x(16*NT) strip; EXEC all-ones (exact grid, no divergence).
// 2-stage software pipeline: prefetch kt+1 fragments while kt's WMMAs execute.
template <int NT, bool OUT_BF16>
__global__ __launch_bounds__(256) void wmma_gemm_bias(const bf16_t* __restrict__ A,
    const bf16_t* __restrict__ B, const float* __restrict__ bias,
    void* __restrict__ out, int M, int N, int K) {
  const int lane = threadIdx.x & 31;
  const int wave = blockIdx.x * 8 + (threadIdx.x >> 5);
  const int Mt = M >> 4;
  const int mt = wave % Mt;
  const int ng = wave / Mt;
  const int m0 = mt << 4;
  const int nt0 = ng * NT;
  const int KT = K >> 5;

  v8f acc[NT] = {};

  const bf16_t* arow  = A + (size_t)(m0 + (lane & 15)) * K + ((lane >> 4) << 3);
  const bf16_t* bbase = B + (((size_t)nt0 * KT) << 9) + (lane << 4);

  v16bf aCur = wmma_load_a(arow);
  v16bf bCur[NT];
#pragma unroll
  for (int i = 0; i < NT; ++i)
    bCur[i] = *(const v16bf*)(bbase + (((size_t)i * KT) << 9));

#pragma unroll 2
  for (int kt = 0; kt < KT; ++kt) {
    const int kn = (kt + 1) & (KT - 1);    // wrap: tail prefetch is L2-hot, keeps EXEC uniform
    v16bf aNxt = wmma_load_a(arow + (kn << 5));
    v16bf bNxt[NT];
#pragma unroll
    for (int i = 0; i < NT; ++i)
      bNxt[i] = *(const v16bf*)(bbase + (((size_t)i * KT + kn) << 9));
#pragma unroll
    for (int i = 0; i < NT; ++i)
      acc[i] = __builtin_amdgcn_wmma_f32_16x16x32_bf16(false, aCur, false, bCur[i],
                                                       (short)0, acc[i], false, false);
    // pin schedule: (2 + 2*NT) VMEM reads first, then NT WMMAs, per iteration
    __builtin_amdgcn_sched_group_barrier(0x020, 2 + 2 * NT, 0);  // VMEM read
    __builtin_amdgcn_sched_group_barrier(0x008, NT, 0);          // MFMA/WMMA
    aCur = aNxt;
#pragma unroll
    for (int i = 0; i < NT; ++i) bCur[i] = bNxt[i];
  }

  // C/D layout: lane holds col n0+(lane&15); VGPR r -> row m0 + r + (lane>=16 ? 8 : 0)
  const int col0  = lane & 15;
  const int rbase = m0 + ((lane >> 4) << 3);
#pragma unroll
  for (int i = 0; i < NT; ++i) {
    const int n = ((nt0 + i) << 4) + col0;
    const float bv = bias[n];
#pragma unroll
    for (int r = 0; r < 8; ++r) {
      const float v = acc[i][r] + bv;
      const size_t off = (size_t)(rbase + r) * N + n;
      if (OUT_BF16) ((bf16_t*)out)[off] = (bf16_t)v;
      else          ((float*)out)[off]  = v;
    }
  }
}

// ---- one LSTM timestep: gates = Xg_t + Hprev @ Wh (4 gates fused per wave) ----
// wave -> (mt in 0..3, ct in 0..63); computes 16x16 col block for gates i,f,o,c
__global__ __launch_bounds__(256) void lstm_step_kernel(const bf16_t* __restrict__ Hprev,
    bf16_t* __restrict__ Hnext, const bf16_t* __restrict__ pWh,
    const bf16_t* __restrict__ Xg_t, const float* __restrict__ Cin,
    float* __restrict__ Cout, float* __restrict__ Hf) {
  const int lane = threadIdx.x & 31;
  const int wave = blockIdx.x * 8 + (threadIdx.x >> 5);
  const int mt = wave & 3;
  const int ct = wave >> 2;                 // 0..63
  const int m0 = mt << 4;
  const int col   = (ct << 4) + (lane & 15);
  const int rbase = m0 + ((lane >> 4) << 3);
  const int KT = 32;                        // K = 1024

  v8f acc[4];
#pragma unroll
  for (int g = 0; g < 4; ++g)
#pragma unroll
    for (int r = 0; r < 8; ++r)
      acc[g][r] = (float)Xg_t[(size_t)(rbase + r) * 4096 + (g << 10) + col];

  const bf16_t* arow  = Hprev + (size_t)(m0 + (lane & 15)) * 1024 + ((lane >> 4) << 3);
  const bf16_t* bbase = pWh + (((size_t)ct * KT) << 9) + (lane << 4);

  v16bf aCur = wmma_load_a(arow);
  v16bf bCur[4];
#pragma unroll
  for (int g = 0; g < 4; ++g)
    bCur[g] = *(const v16bf*)(bbase + ((((size_t)(g << 6)) * KT) << 9));

#pragma unroll 2
  for (int kt = 0; kt < KT; ++kt) {
    const int kn = (kt + 1) & (KT - 1);
    v16bf aNxt = wmma_load_a(arow + (kn << 5));
    v16bf bNxt[4];
#pragma unroll
    for (int g = 0; g < 4; ++g)
      bNxt[g] = *(const v16bf*)(bbase + ((((size_t)(g << 6)) * KT + kn) << 9));
#pragma unroll
    for (int g = 0; g < 4; ++g)
      acc[g] = __builtin_amdgcn_wmma_f32_16x16x32_bf16(false, aCur, false, bCur[g],
                                                       (short)0, acc[g], false, false);
    __builtin_amdgcn_sched_group_barrier(0x020, 10, 0);  // 10 VMEM reads (2 A + 8 B)
    __builtin_amdgcn_sched_group_barrier(0x008, 4, 0);   // 4 WMMA
    aCur = aNxt;
#pragma unroll
    for (int g = 0; g < 4; ++g) bCur[g] = bNxt[g];
  }

#pragma unroll
  for (int r = 0; r < 8; ++r) {
    const int row = rbase + r;
    const size_t off = (size_t)row * 1024 + col;
    const float I  = fast_sigmoid(acc[0][r]);
    const float F  = fast_sigmoid(acc[1][r]);
    const float O  = fast_sigmoid(acc[2][r]);
    const float Ct = fast_tanh(acc[3][r]);
    const float Cn = F * Cin[off] + I * Ct;
    const float Hn = O * fast_tanh(Cn);
    Cout[off]  = Cn;
    Hf[off]    = Hn;
    Hnext[off] = (bf16_t)Hn;
  }
}

extern "C" void kernel_launch(void* const* d_in, const int* in_sizes, int n_in,
                              void* d_out, int out_size, void* d_ws, size_t ws_size,
                              hipStream_t stream) {
  (void)in_sizes; (void)n_in; (void)out_size; (void)ws_size;
  constexpr int T = 256, Bz = 64, D = 1024, Hh = 1024;

  const float* inputs = (const float*)d_in[0];
  const float* H0   = (const float*)d_in[1];
  const float* C0   = (const float*)d_in[2];
  const float* W_xi = (const float*)d_in[3];
  const float* W_hi = (const float*)d_in[4];
  const float* b_i  = (const float*)d_in[5];
  const float* W_xf = (const float*)d_in[6];
  const float* W_hf = (const float*)d_in[7];
  const float* b_f  = (const float*)d_in[8];
  const float* W_xo = (const float*)d_in[9];
  const float* W_ho = (const float*)d_in[10];
  const float* b_o  = (const float*)d_in[11];
  const float* W_xc = (const float*)d_in[12];
  const float* W_hc = (const float*)d_in[13];
  const float* b_c  = (const float*)d_in[14];
  const float* W_hq = (const float*)d_in[15];
  const float* b_q  = (const float*)d_in[16];

  // workspace layout (bytes), all 32B-aligned
  char* ws = (char*)d_ws;
  bf16_t* pWx   = (bf16_t*)(ws + 0);                      //  8 MB  packed [1024 x 4096]
  bf16_t* pWh   = (bf16_t*)(ws + (size_t)8  * 1048576);   //  8 MB  packed [1024 x 4096]
  bf16_t* pWq   = (bf16_t*)(ws + (size_t)16 * 1048576);   //  2 MB  packed [1024 x 1024]
  float*  bias4 = (float*) (ws + (size_t)18 * 1048576);   // 16 KB  [4096]
  bf16_t* Xbf   = (bf16_t*)(ws + (size_t)18 * 1048576 + 16384);  // 32 MB  bf16 inputs [16384 x 1024]
  bf16_t* Xg    = (bf16_t*)(ws + (size_t)52445184);       // 128 MB bf16 pre-gates [16384 x 4096]
  bf16_t* Hs    = (bf16_t*)(ws + (size_t)186662912);      // ~34 MB bf16 H history [(T+1) x 64 x 1024]

  float* outF = (float*)d_out;                  // outputs [T,B,D]
  float* Hf   = outF + (size_t)T * Bz * D;      // Hf [B,H]
  float* Cf   = Hf   + (size_t)Bz * Hh;         // Cf [B,H]

  // 1) convert activations to bf16
  f2bf_kernel<<<65536, 256, 0, stream>>>(inputs, Xbf, T * Bz * D);
  f2bf_kernel<<<256,   256, 0, stream>>>(H0, Hs, Bz * Hh);     // Hs[0] = H0

  // 2) pack weights into WMMA B-fragment layout
  pack_b_kernel<<<16384, 256, 0, stream>>>(W_xi, W_xf, W_xo, W_xc, pWx, 1024, 4096);
  pack_b_kernel<<<16384, 256, 0, stream>>>(W_hi, W_hf, W_ho, W_hc, pWh, 1024, 4096);
  pack_b_kernel<<<4096,  256, 0, stream>>>(W_hq, W_hq, W_hq, W_hq, pWq, 1024, 1024);
  concat_bias_kernel<<<16, 256, 0, stream>>>(b_i, b_f, b_o, b_c, bias4);

  // 3) Xg = inputs @ [Wxi|Wxf|Wxo|Wxc] + bias   (M=16384, N=4096, K=1024)
  wmma_gemm_bias<4, true><<<8192, 256, 0, stream>>>(Xbf, pWx, bias4, Xg, 16384, 4096, 1024);

  // 4) sequential recurrent scan (weights stay hot in 192MB L2)
  for (int t = 0; t < T; ++t) {
    const float* Cin = (t == 0) ? C0 : (const float*)Cf;
    lstm_step_kernel<<<32, 256, 0, stream>>>(
        Hs + (size_t)t * Bz * Hh, Hs + (size_t)(t + 1) * Bz * Hh,
        pWh, Xg + (size_t)t * Bz * 4096, Cin, Cf, Hf);
  }

  // 5) outputs = Hs[1..T] @ W_hq + b_q   (M=16384, N=1024, K=1024)
  wmma_gemm_bias<4, false><<<2048, 256, 0, stream>>>(
      Hs + (size_t)Bz * Hh, pWq, b_q, outF, 16384, 1024, 1024);
}
